// CSPNetLight_MessagePassing_11192684773494
// MI455X (gfx1250) — matrix operand order repaired and checked by reading, not compiled, not run on hardware
//
#include <hip/hip_runtime.h>
#include <hip/hip_bf16.h>

#define BSZ 16
#define NN  128
#define HH  256
#define LLD 9
#define FF  48

typedef __attribute__((ext_vector_type(16))) __bf16       bf16x16;
typedef __attribute__((ext_vector_type(8)))  float        f32x8;
typedef __attribute__((ext_vector_type(8)))  unsigned int u32x8;

// Fast SiLU: v_mul + v_exp_f32 + v_add + v_rcp_f32 + v_mul (no IEEE div sequence)
__device__ __forceinline__ float silu_f(float x) {
    float e = __expf(-x);
    return x * __builtin_amdgcn_rcpf(1.0f + e);
}

// ---------------------------------------------------------------------------
// Kernel 1: P[b,j,h] = node[b,j,:] @ Wj ; R[b,j,h] = node[b,j,:] @ Wi + g@Wg + b1
// ---------------------------------------------------------------------------
__global__ __launch_bounds__(256)
void prep_pr(const float* __restrict__ node, const float* __restrict__ graph,
             const float* __restrict__ W1, const float* __restrict__ b1,
             float* __restrict__ P, float* __restrict__ R) {
    int bj = blockIdx.x;              // b*NN + j
    int b  = bj / NN;
    int h  = threadIdx.x;             // 0..255
    __shared__ float nrow[HH];
    nrow[h] = node[(long)bj * HH + h];
    __syncthreads();
    float accP = 0.f, accR = 0.f;
    for (int k = 0; k < HH; ++k) {
        float nk = nrow[k];
        accP += nk * W1[(long)k * HH + h];           // Wj rows 0..255
        accR += nk * W1[(long)(HH + k) * HH + h];    // Wi rows 256..511
    }
    for (int l = 0; l < LLD; ++l)
        accR += graph[b * LLD + l] * W1[(long)(2 * HH + l) * HH + h];
    accR += b1[h];
    P[(long)bj * HH + h] = accP;
    R[(long)bj * HH + h] = accR;
}

// ---------------------------------------------------------------------------
// Kernel 2: bf16 n-major weight copies.
//   WeT[n][kk] (kk<64, zero-padded past 48) from W1 rows 521..568
//   W2T[n][k]  from W2 (HxH row-major)
// ---------------------------------------------------------------------------
__global__ __launch_bounds__(256)
void prep_w(const float* __restrict__ W1, const float* __restrict__ W2,
            __bf16* __restrict__ WeT, __bf16* __restrict__ W2T) {
    int n = blockIdx.x;               // 0..255 (output column)
    int t = threadIdx.x;              // 0..255
    if (t < 64) {
        float v = (t < FF) ? W1[(long)(2 * HH + LLD + t) * HH + n] : 0.0f;
        WeT[n * 64 + t] = (__bf16)v;
    }
    W2T[(long)n * HH + t] = (__bf16)W2[(long)t * HH + n];
}

// ---------------------------------------------------------------------------
// Kernel 3: fused edge MLP + mean over j.  One WG per (b,i); 8 waves, each
// owning a 16-row j-tile. All matmuls via v_wmma_f32_16x16x32_bf16.
// ---------------------------------------------------------------------------
#define SH 264   // ldsH row stride in bf16 (even, padded)

__global__ __launch_bounds__(256)
void edge_mlp(const float* __restrict__ edge,
              const float* __restrict__ P,
              const float* __restrict__ R,
              const __bf16* __restrict__ WeT,
              const __bf16* __restrict__ W2T,
              const float* __restrict__ b2,
              float* __restrict__ out) {
    __shared__ __bf16 ldsA[8][16 * 64];    // per-wave edge tile (bf16, K padded)
    __shared__ __bf16 ldsH[128 * SH];      // H1 (bf16), wave-private 16-row bands
    __shared__ float  out_acc[HH];
    __shared__ float  rrow[HH];

    const int blk  = blockIdx.x;           // b*NN + i
    const int b    = blk / NN;
    const int tid  = threadIdx.x;
    const int w    = tid >> 5;             // wave id 0..7  (j-tile)
    const int lane = tid & 31;
    const int half = lane >> 4;
    const int ln   = lane & 15;

    rrow[tid]    = R[(long)blk * HH + tid];   // pi + pg + b1 for this (b,i)
    out_acc[tid] = 0.0f;

    // ---- stage this wave's 16x48 edge rows into LDS as bf16, pad K to 64
    const float* erow = edge + ((long)blk * NN + w * 16) * FF;
    for (int idx = lane; idx < 16 * FF; idx += 32) {
        int r = idx / FF, f = idx - r * FF;
        ldsA[w][r * 64 + f] = (__bf16)erow[r * FF + f];
    }
    for (int idx = lane; idx < 16 * 16; idx += 32) {
        int r = idx >> 4, f = idx & 15;
        ldsA[w][r * 64 + 48 + f] = (__bf16)0.0f;
    }
    __syncthreads();

    // ---- GEMM1: Z = edge(16x64) @ We(64x256); H1 = silu(Z + P + R) -> ldsH
    const unsigned* weu = (const unsigned*)WeT;   // uint = 2 bf16 along K
    for (int nt = 0; nt < 16; ++nt) {
        f32x8 acc = {};
#pragma unroll
        for (int ks = 0; ks < 2; ++ks) {
            u32x8 au, bu;
            const unsigned* arow =
                (const unsigned*)&ldsA[w][ln * 64 + ks * 32];
#pragma unroll
            for (int v = 0; v < 8; ++v) {
                int kb = ((v & 4) << 2) + half * 8 + (v & 3) * 2; // A layout
                au[v] = arow[kb >> 1];
            }
            const unsigned* brow =
                weu + (nt * 16 + ln) * 32 + ks * 16 + half * 4;   // B layout
#pragma unroll
            for (int v = 0; v < 8; ++v) bu[v] = brow[((v & 4) << 1) + (v & 3)];
            bf16x16 af = __builtin_bit_cast(bf16x16, au);
            bf16x16 bf = __builtin_bit_cast(bf16x16, bu);
            acc = __builtin_amdgcn_wmma_f32_16x16x32_bf16(
                false, af, false, bf, (short)0, acc, false, false);
        }
        int c = nt * 16 + ln;
        float rc = rrow[c];
        const float* prow = P + ((long)b * NN + w * 16) * HH + c;
#pragma unroll
        for (int v = 0; v < 8; ++v) {
            int m = v + half * 8;                   // C/D layout row
            float z = acc[v] + prow[(long)m * HH] + rc;
            ldsH[(w * 16 + m) * SH + c] = (__bf16)silu_f(z);
        }
    }
    // ldsH band is wave-private: same-wave ds ordering (s_wait_dscnt) suffices.

    // ---- GEMM2: E = silu(H1(16x256) @ W2(256x256) + b2); column partial sums
    // Hoist all A fragments for this wave's 16-row band: 8 K-steps x 8 dwords.
    u32x8 aall[8];
#pragma unroll
    for (int ks = 0; ks < 8; ++ks) {
        const unsigned* arow =
            (const unsigned*)&ldsH[(w * 16 + ln) * SH + ks * 32];
#pragma unroll
        for (int v = 0; v < 8; ++v) {
            int kb = ((v & 4) << 2) + half * 8 + (v & 3) * 2;
            aall[ks][v] = arow[kb >> 1];
        }
    }

    const unsigned* w2u = (const unsigned*)W2T;
#pragma unroll 2
    for (int nt = 0; nt < 16; ++nt) {
        f32x8 acc = {};
#pragma unroll
        for (int ks = 0; ks < 8; ++ks) {
            u32x8 bu;
            const unsigned* brow =
                w2u + (nt * 16 + ln) * 128 + ks * 16 + half * 4;
#pragma unroll
            for (int v = 0; v < 8; ++v) bu[v] = brow[((v & 4) << 1) + (v & 3)];
            bf16x16 af = __builtin_bit_cast(bf16x16, aall[ks]);
            bf16x16 bf = __builtin_bit_cast(bf16x16, bu);
            acc = __builtin_amdgcn_wmma_f32_16x16x32_bf16(
                false, af, false, bf, (short)0, acc, false, false);
        }
        int c = nt * 16 + ln;
        float bias = b2[c];
        float psum = 0.0f;
#pragma unroll
        for (int v = 0; v < 8; ++v) psum += silu_f(acc[v] + bias);
        atomicAdd(&out_acc[c], psum);               // ds_add_f32
    }
    __syncthreads();
    out[(long)blk * HH + tid] = out_acc[tid] * (1.0f / (float)NN);
}

// ---------------------------------------------------------------------------
extern "C" void kernel_launch(void* const* d_in, const int* in_sizes, int n_in,
                              void* d_out, int out_size, void* d_ws, size_t ws_size,
                              hipStream_t stream) {
    const float* node  = (const float*)d_in[0];   // (16,128,256)
    const float* edge  = (const float*)d_in[1];   // (16,128,128,48)
    const float* graph = (const float*)d_in[2];   // (16,9)
    const float* W1    = (const float*)d_in[3];   // (569,256)
    const float* b1    = (const float*)d_in[4];   // (256,)
    const float* W2    = (const float*)d_in[5];   // (256,256)
    const float* b2    = (const float*)d_in[6];   // (256,)
    float*       out   = (float*)d_out;           // (16,128,256)

    // workspace layout
    float*  P   = (float*)d_ws;                       // 16*128*256 f32
    float*  R   = P + (long)BSZ * NN * HH;            // 16*128*256 f32
    __bf16* WeT = (__bf16*)(R + (long)BSZ * NN * HH); // 256*64 bf16
    __bf16* W2T = WeT + 256 * 64;                     // 256*256 bf16

    prep_pr<<<BSZ * NN, 256, 0, stream>>>(node, graph, W1, b1, P, R);
    prep_w<<<HH, 256, 0, stream>>>(W1, W2, WeT, W2T);
    edge_mlp<<<BSZ * NN, 256, 0, stream>>>(edge, P, R, WeT, W2T, b2, out);
}
